// GAT_GCN_601295421671
// MI455X (gfx1250) — matrix operand order repaired and checked by reading, not compiled, not run on hardware
//
#include <hip/hip_runtime.h>

// ---------------- model dims ----------------
#define NN     16384
#define EE     65536
#define E2     (EE + NN)      // edges + self loops
#define BB     256
#define NPG    64
#define FXD    78
#define HEADS  10
#define DG     780            // FXD*HEADS
#define LSMI   200
#define TPROT  512
#define HGRU   100
#define HL     32             // LSTM hidden

// ---------------- WMMA / TDM types ----------------
typedef __attribute__((ext_vector_type(16))) __bf16 v16bf;
typedef __attribute__((ext_vector_type(8)))  __bf16 v8bf;
typedef __attribute__((ext_vector_type(8)))  float  v8f;
typedef __attribute__((ext_vector_type(4)))  unsigned v4u;
typedef __attribute__((ext_vector_type(8)))  int      v8i;
typedef __attribute__((ext_vector_type(4)))  int      v4i;

#if defined(__gfx1250__) && __has_builtin(__builtin_amdgcn_tensor_load_to_lds)
#define HAVE_TDM 1
#else
#define HAVE_TDM 0
#endif

// ---------------- small device helpers ----------------
static __device__ __forceinline__ float sigm(float x) { return 1.f / (1.f + __expf(-x)); }
static __device__ __forceinline__ float lrelu(float x) { return x > 0.f ? x : 0.2f * x; }

static __device__ __forceinline__ unsigned short f2bf(float f) {
    unsigned int x = __float_as_uint(f);
    unsigned int r = x + 0x7FFFu + ((x >> 16) & 1u);   // RNE
    return (unsigned short)(r >> 16);
}
// order-preserving float<->uint key for atomicMax-based segment max
static __device__ __forceinline__ unsigned fkey(float f) {
    unsigned b = __float_as_uint(f);
    unsigned mask = (b >> 31) ? 0xFFFFFFFFu : 0x80000000u;
    return b ^ mask;
}
static __device__ __forceinline__ float fkey_dec(unsigned k) {
    unsigned b = (k & 0x80000000u) ? (k ^ 0x80000000u) : ~k;
    return __uint_as_float(b);
}
static __device__ __forceinline__ void edge_sd(const int* __restrict__ ei, int e, int& s, int& d) {
    if (e < EE) { s = ei[e]; d = ei[EE + e]; } else { s = e - EE; d = e - EE; }
}

#if HAVE_TDM
// Issue a TDM DMA of a [rows=128 x cols=32] bf16 tile (row stride = KpElems) from
// global to LDS at byte offset ldsOff.  D# layout per CDNA5 ISA §8.3/8.4.
static __device__ __forceinline__ void tdm_tile(const unsigned short* gtile,
                                                unsigned ldsOff, int KpElems,
                                                int rows, int cols) {
    unsigned long long ga = (unsigned long long)(size_t)gtile;
    v4u g0 = { 1u,                                    // count=1 (valid, user mode)
               ldsOff,                                // lds_addr
               (unsigned)(ga & 0xFFFFFFFFu),          // global_addr[31:0]
               ((unsigned)((ga >> 32) & 0x01FFFFFFu)) | (2u << 30) };  // addr[56:32] | type=2
    v8i g1 = { (int)(1u << 16),                       // data_size=1 -> 2 bytes; wg_mask=0
               (int)(0xFFFFu << 16),                  // tensor_dim0[15:0]=0xFFFF (huge, no OOB)
               (int)(0x7FFFu | (0xFFFFu << 16)),      // tensor_dim0[31:16] | tensor_dim1[15:0]
               (int)(0x7FFFu | ((unsigned)cols << 16)), // tensor_dim1[31:16] | tile_dim0
               (int)rows,                             // tile_dim1 | tile_dim2=0
               KpElems,                               // tensor_dim0_stride[31:0] (elements)
               0, 0 };
    v4i gz = { 0, 0, 0, 0 };
#if __clang_major__ >= 23
    v8i gz8 = { 0, 0, 0, 0, 0, 0, 0, 0 };
    __builtin_amdgcn_tensor_load_to_lds(g0, g1, gz, gz, gz8, 0);
#else
    __builtin_amdgcn_tensor_load_to_lds(g0, g1, gz, gz, 0);
#endif
}
#endif

// ---------------- generic utility kernels ----------------
__global__ void k_fill_u32(unsigned* __restrict__ p, unsigned v, long long n) {
    long long i = (long long)blockIdx.x * blockDim.x + threadIdx.x;
    if (i < n) p[i] = v;
}
// A staging: src f32 [M,K] -> dst bf16 [Mp,Kp], zero padded.  (also used for B^T when
// the torch weight is (out,in) row-major, which is exactly B^T = [N,K])
__global__ void k_stage_a(const float* __restrict__ s, unsigned short* __restrict__ d,
                          int M, int K, int Kp, long long total) {
    long long i = (long long)blockIdx.x * blockDim.x + threadIdx.x;
    if (i >= total) return;
    int m = (int)(i / Kp), k = (int)(i % Kp);
    d[i] = (m < M && k < K) ? f2bf(s[(size_t)m * K + k]) : (unsigned short)0;
}
// B^T staging from a [K,N] row-major matrix: dst bf16 [Nbp,Kp] with dst[n,k]=src[k,n]
__global__ void k_stage_bt_trans(const float* __restrict__ s, unsigned short* __restrict__ d,
                                 int N, int K, int Kp, long long total) {
    long long i = (long long)blockIdx.x * blockDim.x + threadIdx.x;
    if (i >= total) return;
    int n = (int)(i / Kp), k = (int)(i % Kp);
    d[i] = (n < N && k < K) ? f2bf(s[(size_t)k * N + n]) : (unsigned short)0;
}

// ---------------- WMMA GEMM ----------------
// C[M,N](f32) = act(A[Mp,Kp]bf16 * B^T[Nbp,Kp]bf16 + bias)
// Mp,Nbp multiples of 128; Kp multiple of 32; padding is zero-filled.
#define BM 128
#define BN 128
#define BK 32

__global__ __launch_bounds__(256) void k_gemm_bf16(
    const unsigned short* __restrict__ A,   // [Mp, Kp]
    const unsigned short* __restrict__ Bt,  // [Nbp, Kp]
    const float* __restrict__ bias, float* __restrict__ C,
    int M, int N, int Kp, int ldc, int act, int has_bias)
{
    __shared__ alignas(16) unsigned short As[2][BM * BK];
    __shared__ alignas(16) unsigned short Bs[2][BN * BK];
    int tid  = threadIdx.x;
    int lane = tid & 31;
    int w    = tid >> 5;       // 8 waves
    int wm   = w >> 1;         // 0..3 -> 32-row strip
    int wn   = w & 1;          // 0..1 -> 64-col strip
    int m0 = blockIdx.y * BM;
    int n0 = blockIdx.x * BN;
    int nCh = Kp / BK;

    v8f acc[2][4];
#pragma unroll
    for (int i = 0; i < 2; ++i)
#pragma unroll
        for (int j = 0; j < 4; ++j)
#pragma unroll
            for (int e = 0; e < 8; ++e) acc[i][j][e] = 0.f;

#if HAVE_TDM
    // TDM feeds the tiles: wave0 DMAs next chunk while all waves run WMMA.
    unsigned asOff = (unsigned)(size_t)&As[0][0];
    unsigned bsOff = (unsigned)(size_t)&Bs[0][0];
    const unsigned bufBytes = BM * BK * 2;   // 8 KiB per buffer
    if (w == 0) {
        tdm_tile(&A [(size_t)m0 * Kp], asOff, Kp, BM, BK);
        tdm_tile(&Bt[(size_t)n0 * Kp], bsOff, Kp, BN, BK);
        __builtin_amdgcn_s_wait_tensorcnt(0);
    }
    __syncthreads();
#else
    // cooperative tile load: each thread owns 16 contiguous shorts (2x uint4)
    int r  = tid >> 1;              // 0..127
    int kk = (tid & 1) * 16;        // 0 or 16
    const uint4* gA = (const uint4*)&A [(size_t)(m0 + r) * Kp + kk];
    const uint4* gB = (const uint4*)&Bt[(size_t)(n0 + r) * Kp + kk];
    int sOff = r * BK + kk;
    uint4 pa0 = gA[0], pa1 = gA[1];
    uint4 pb0 = gB[0], pb1 = gB[1];
    { uint4* s = (uint4*)&As[0][sOff]; s[0] = pa0; s[1] = pa1;
      uint4* t = (uint4*)&Bs[0][sOff]; t[0] = pb0; t[1] = pb1; }
    __syncthreads();
#endif

    int half  = lane >> 4;
    int lan15 = lane & 15;
    for (int ci = 0; ci < nCh; ++ci) {
        int cur = ci & 1, nxt = cur ^ 1;
#if HAVE_TDM
        if (ci + 1 < nCh && w == 0) {
            tdm_tile(&A [(size_t)m0 * Kp + (size_t)(ci + 1) * BK], asOff + nxt * bufBytes, Kp, BM, BK);
            tdm_tile(&Bt[(size_t)n0 * Kp + (size_t)(ci + 1) * BK], bsOff + nxt * bufBytes, Kp, BN, BK);
        }
#else
        if (ci + 1 < nCh) {
            const uint4* ga = gA + (size_t)(ci + 1) * 4;
            const uint4* gb = gB + (size_t)(ci + 1) * 4;
            pa0 = ga[0]; pa1 = ga[1];
            pb0 = gb[0]; pb1 = gb[1];
            if (ci + 2 < nCh) {
                __builtin_prefetch(gA + (size_t)(ci + 2) * 4, 0, 1);  // global_prefetch_b8
                __builtin_prefetch(gB + (size_t)(ci + 2) * 4, 0, 1);
            }
        }
#endif
        const unsigned short* Ab = &As[cur][0];
        const unsigned short* Bb = &Bs[cur][0];
        v16bf af[2], bfr[4];
#pragma unroll
        for (int i = 0; i < 2; ++i) {
            int rr = wm * 32 + i * 16 + lan15;
            v8bf lo = *(const v8bf*)&Ab[rr * BK + 8 * half];        // K = 8h .. 8h+7
            v8bf hi = *(const v8bf*)&Ab[rr * BK + 16 + 8 * half];   // K = 16+8h ..
            af[i] = __builtin_shufflevector(lo, hi, 0,1,2,3,4,5,6,7,8,9,10,11,12,13,14,15);
        }
#pragma unroll
        for (int j = 0; j < 4; ++j) {
            int nn = wn * 64 + j * 16 + lan15;
            v8bf lo = *(const v8bf*)&Bb[nn * BK + 8 * half];        // K = 8h .. 8h+7
            v8bf hi = *(const v8bf*)&Bb[nn * BK + 16 + 8 * half];
            bfr[j] = __builtin_shufflevector(lo, hi, 0,1,2,3,4,5,6,7,8,9,10,11,12,13,14,15);
        }
#pragma unroll
        for (int i = 0; i < 2; ++i)
#pragma unroll
            for (int j = 0; j < 4; ++j)
                acc[i][j] = __builtin_amdgcn_wmma_f32_16x16x32_bf16(
                    false, af[i], false, bfr[j], (short)0, acc[i][j], false, false);
        __syncthreads();
        if (ci + 1 < nCh) {
#if HAVE_TDM
            if (w == 0) __builtin_amdgcn_s_wait_tensorcnt(0);
            __syncthreads();
#else
            uint4* s = (uint4*)&As[nxt][sOff]; s[0] = pa0; s[1] = pa1;
            uint4* t = (uint4*)&Bs[nxt][sOff]; t[0] = pb0; t[1] = pb1;
            __syncthreads();
#endif
        }
    }
    // epilogue: C/D layout -> VGPR e: row = base + e + 8*half, col = base + lane15
#pragma unroll
    for (int i = 0; i < 2; ++i)
#pragma unroll
        for (int j = 0; j < 4; ++j) {
            int rbase = m0 + wm * 32 + i * 16;
            int col   = n0 + wn * 64 + j * 16 + lan15;
#pragma unroll
            for (int e = 0; e < 8; ++e) {
                int row = rbase + e + 8 * half;
                if (row < M && col < N) {
                    float v = acc[i][j][e];
                    if (has_bias) v += bias[col];
                    if (act) v = fmaxf(v, 0.f);
                    C[(size_t)row * ldc + col] = v;
                }
            }
        }
}

// ---------------- GAT pieces ----------------
__global__ void k_gat_scores(const float* __restrict__ h, const float* __restrict__ asrc,
                             const float* __restrict__ adst, float* __restrict__ a_s, float* __restrict__ a_d) {
    int idx = blockIdx.x * blockDim.x + threadIdx.x;
    if (idx >= NN * HEADS) return;
    int node = idx / HEADS, hh = idx % HEADS;
    const float* hp = h + (size_t)node * DG + hh * FXD;
    float ss = 0.f, sd = 0.f;
    for (int k = 0; k < FXD; ++k) { float v = hp[k]; ss += v * asrc[hh * FXD + k]; sd += v * adst[hh * FXD + k]; }
    a_s[idx] = ss; a_d[idx] = sd;
}
__global__ void k_edge_max(const int* __restrict__ ei, const float* __restrict__ a_s,
                           const float* __restrict__ a_d, unsigned* __restrict__ mkey) {
    int idx = blockIdx.x * blockDim.x + threadIdx.x;
    if (idx >= E2 * HEADS) return;
    int e = idx / HEADS, hh = idx % HEADS, s, d;
    edge_sd(ei, e, s, d);
    float lg = lrelu(a_s[s * HEADS + hh] + a_d[d * HEADS + hh]);
    atomicMax(&mkey[d * HEADS + hh], fkey(lg));
}
__global__ void k_edge_exp(const int* __restrict__ ei, const float* __restrict__ a_s,
                           const float* __restrict__ a_d, const unsigned* __restrict__ mkey,
                           float* __restrict__ ebuf, float* __restrict__ denom) {
    int idx = blockIdx.x * blockDim.x + threadIdx.x;
    if (idx >= E2 * HEADS) return;
    int e = idx / HEADS, hh = idx % HEADS, s, d;
    edge_sd(ei, e, s, d);
    float lg = lrelu(a_s[s * HEADS + hh] + a_d[d * HEADS + hh]);
    float ex = __expf(lg - fkey_dec(mkey[d * HEADS + hh]));
    ebuf[idx] = ex;
    atomicAdd(&denom[d * HEADS + hh], ex);
}
__global__ void k_edge_alpha(const int* __restrict__ ei, const float* __restrict__ denom,
                             float* __restrict__ ebuf) {
    int idx = blockIdx.x * blockDim.x + threadIdx.x;
    if (idx >= E2 * HEADS) return;
    int e = idx / HEADS, hh = idx % HEADS, s, d;
    edge_sd(ei, e, s, d);
    ebuf[idx] = ebuf[idx] / (denom[d * HEADS + hh] + 1e-16f);
}
// one block per edge, threadIdx.x = feature (no div/mod, no exec juggling)
__global__ __launch_bounds__(96) void k_gat_aggr(const int* __restrict__ ei, const float* __restrict__ h,
                                                 const float* __restrict__ alpha, float* __restrict__ out) {
    int e = blockIdx.x;
    int k = threadIdx.x;
    if (k >= FXD) return;
    int s, d; edge_sd(ei, e, s, d);
    const float* hp = h + (size_t)s * DG + k;
    float* op = out + (size_t)d * DG + k;
    const float* ap = alpha + (size_t)e * HEADS;
#pragma unroll
    for (int hh = 0; hh < HEADS; ++hh)
        atomicAdd(op + hh * FXD, hp[hh * FXD] * ap[hh]);
}
__global__ void k_bias_relu(float* __restrict__ x, const float* __restrict__ b, long long n, int cols) {
    long long i = (long long)blockIdx.x * blockDim.x + threadIdx.x;
    if (i < n) x[i] = fmaxf(x[i] + b[i % cols], 0.f);
}
// ---------------- GCN pieces ----------------
__global__ void k_deg(const int* __restrict__ ei, float* __restrict__ deg) {
    int e = blockIdx.x * blockDim.x + threadIdx.x;
    if (e >= E2) return;
    int s, d; edge_sd(ei, e, s, d);
    atomicAdd(&deg[d], 1.f);
}
__global__ void k_dinv(const float* __restrict__ deg, float* __restrict__ dinv) {
    int i = blockIdx.x * blockDim.x + threadIdx.x;
    if (i < NN) dinv[i] = rsqrtf(deg[i]);
}
// grid = (E2, cdiv(DG,256)); one scaled atomicAdd per thread
__global__ __launch_bounds__(256) void k_gcn_aggr(const int* __restrict__ ei, const float* __restrict__ hg,
                                                  const float* __restrict__ dinv, float* __restrict__ out) {
    int e = blockIdx.x;
    int k = blockIdx.y * 256 + threadIdx.x;
    if (k >= DG) return;
    int s, d; edge_sd(ei, e, s, d);
    atomicAdd(&out[(size_t)d * DG + k], hg[(size_t)s * DG + k] * (dinv[s] * dinv[d]));
}
// ---------------- pooling ----------------
__global__ __launch_bounds__(256) void k_pool(const float* __restrict__ xg, float* __restrict__ pooled) {
    int b = blockIdx.x, tid = threadIdx.x;
    for (int f = tid; f < DG; f += 256) {
        float mx = -1e30f, sm = 0.f;
        for (int j = 0; j < NPG; ++j) {
            float v = xg[((size_t)b * NPG + j) * DG + f];
            mx = fmaxf(mx, v); sm += v;
        }
        pooled[(size_t)b * (2 * DG) + f]      = mx;
        pooled[(size_t)b * (2 * DG) + DG + f] = sm * (1.f / NPG);
    }
}
// ---------------- GRU persistent scan (16 batch rows / block) ----------------
#define GR 16
__global__ __launch_bounds__(256) void k_gru_scan(
    const int* __restrict__ smile, const float* __restrict__ emb,
    const float* __restrict__ wi, const float* __restrict__ wh,
    const float* __restrict__ bi, const float* __restrict__ bh,
    float* __restrict__ xc, int dir)
{
    __shared__ float sx[GR][HGRU];
    __shared__ float sh[GR][HGRU];
    __shared__ float sgx[GR][3 * HGRU];
    __shared__ float sgh[GR][3 * HGRU];
    __shared__ float sacc[GR][HGRU];
    int tid = threadIdx.x;
    int b0 = blockIdx.x * GR;
    for (int i = tid; i < GR * HGRU; i += 256) { sh[i / HGRU][i % HGRU] = 0.f; sacc[i / HGRU][i % HGRU] = 0.f; }
    __syncthreads();
    for (int ts = 0; ts < LSMI; ++ts) {
        int t = dir ? (LSMI - 1 - ts) : ts;
        for (int i = tid; i < GR * HGRU; i += 256) {
            int r = i / HGRU, k = i % HGRU;
            int tok = smile[(b0 + r) * LSMI + t];
            sx[r][k] = emb[tok * HGRU + k];
        }
        __syncthreads();
        for (int i = tid; i < GR * 3 * HGRU; i += 256) {
            int r = i / (3 * HGRU), j = i % (3 * HGRU);
            float gx = bi[j], gh = bh[j];
            const float* wij = wi + j * HGRU;
            const float* whj = wh + j * HGRU;
            for (int k = 0; k < HGRU; ++k) { gx += sx[r][k] * wij[k]; gh += sh[r][k] * whj[k]; }
            sgx[r][j] = gx; sgh[r][j] = gh;
        }
        __syncthreads();
        for (int i = tid; i < GR * HGRU; i += 256) {
            int r = i / HGRU, c = i % HGRU;
            float rr = sigm(sgx[r][c] + sgh[r][c]);
            float zz = sigm(sgx[r][HGRU + c] + sgh[r][HGRU + c]);
            float nn = tanhf(sgx[r][2 * HGRU + c] + rr * sgh[r][2 * HGRU + c]);
            float h2 = (1.f - zz) * nn + zz * sh[r][c];
            sh[r][c] = h2;
            sacc[r][c] += fmaxf(h2, 0.f);
        }
        __syncthreads();
    }
    for (int i = tid; i < GR * HGRU; i += 256) {
        int r = i / HGRU, c = i % HGRU;
        xc[(size_t)(b0 + r) * 520 + 128 + dir * HGRU + c] = sacc[r][c] * (1.f / LSMI);
    }
}
// ---------------- LSTM persistent scan ----------------
#define LR 16
__global__ __launch_bounds__(256) void k_lstm_scan(
    const float* __restrict__ yin, const int* __restrict__ target, const float* __restrict__ embxt,
    const float* __restrict__ wi, const float* __restrict__ wh,
    const float* __restrict__ bi, const float* __restrict__ bh,
    float* __restrict__ yout, float* __restrict__ hn,
    int in_dim, int layer0, int dir, int hn_idx)
{
    __shared__ float sx[LR][128];
    __shared__ float sg[LR][4 * HL];
    __shared__ float sh[LR][HL];
    __shared__ float sc[LR][HL];
    int tid = threadIdx.x;
    int b0 = blockIdx.x * LR;
    for (int i = tid; i < LR * HL; i += 256) { sh[i / HL][i % HL] = 0.f; sc[i / HL][i % HL] = 0.f; }
    __syncthreads();
    for (int ts = 0; ts < TPROT; ++ts) {
        int t = dir ? (TPROT - 1 - ts) : ts;
        for (int i = tid; i < LR * in_dim; i += 256) {
            int r = i / in_dim, k = i % in_dim;
            float v;
            if (layer0) { int tok = target[(b0 + r) * TPROT + t]; v = embxt[tok * 128 + k]; }
            else        { v = yin[((size_t)t * BB + (b0 + r)) * 64 + k]; }
            sx[r][k] = v;
        }
        __syncthreads();
        for (int i = tid; i < LR * 4 * HL; i += 256) {
            int r = i >> 7, j = i & 127;
            float s = bi[j] + bh[j];
            const float* wij = wi + j * in_dim;
            for (int k = 0; k < in_dim; ++k) s += sx[r][k] * wij[k];
            const float* whj = wh + j * HL;
            for (int k = 0; k < HL; ++k) s += sh[r][k] * whj[k];
            sg[r][j] = s;
        }
        __syncthreads();
        for (int i = tid; i < LR * HL; i += 256) {
            int r = i >> 5, k = i & 31;
            float ig = sigm(sg[r][k]);
            float fg = sigm(sg[r][HL + k]);
            float gg = tanhf(sg[r][2 * HL + k]);
            float og = sigm(sg[r][3 * HL + k]);
            float c2 = fg * sc[r][k] + ig * gg;
            float h2 = og * tanhf(c2);
            sc[r][k] = c2; sh[r][k] = h2;
            yout[((size_t)t * BB + (b0 + r)) * 64 + dir * HL + k] = h2;
        }
        __syncthreads();
    }
    for (int i = tid; i < LR * HL; i += 256) {
        int r = i >> 5, k = i & 31;
        hn[(size_t)hn_idx * BB * HL + (b0 + r) * HL + k] = sh[r][k];
    }
}
// ---------------- attention over LSTM outputs ----------------
__global__ __launch_bounds__(256) void k_attention(const float* __restrict__ y,
                                                   const float* __restrict__ hn,
                                                   float* __restrict__ xc) {
    int b = blockIdx.x, tid = threadIdx.x;
    __shared__ float hid[64 * 3];
    __shared__ float attn[TPROT * 3];
    __shared__ float red[256];
    for (int i = tid; i < 192; i += 256) hid[i] = hn[b * 192 + i];   // "faithful" flat reshape
    __syncthreads();
    for (int t = tid; t < TPROT; t += 256) {
        const float* yt = y + ((size_t)t * BB + b) * 64;
        float a0 = 0.f, a1 = 0.f, a2 = 0.f;
        for (int k = 0; k < 64; ++k) { float v = yt[k]; a0 += v * hid[k * 3]; a1 += v * hid[k * 3 + 1]; a2 += v * hid[k * 3 + 2]; }
        attn[t * 3] = a0; attn[t * 3 + 1] = a1; attn[t * 3 + 2] = a2;
    }
    __syncthreads();
    for (int c = 0; c < 3; ++c) {
        float mx = -1e30f;
        for (int t = tid; t < TPROT; t += 256) mx = fmaxf(mx, attn[t * 3 + c]);
        red[tid] = mx; __syncthreads();
        for (int s = 128; s > 0; s >>= 1) { if (tid < s) red[tid] = fmaxf(red[tid], red[tid + s]); __syncthreads(); }
        mx = red[0]; __syncthreads();
        float sm = 0.f;
        for (int t = tid; t < TPROT; t += 256) { float e = __expf(attn[t * 3 + c] - mx); attn[t * 3 + c] = e; sm += e; }
        red[tid] = sm; __syncthreads();
        for (int s = 128; s > 0; s >>= 1) { if (tid < s) red[tid] += red[tid + s]; __syncthreads(); }
        sm = red[0]; __syncthreads();
        float inv = 1.f / sm;
        for (int t = tid; t < TPROT; t += 256) attn[t * 3 + c] *= inv;
        __syncthreads();
    }
    for (int i = tid; i < 192; i += 256) {
        int k = i / 3, c = i % 3;
        float s = 0.f;
        for (int t = 0; t < TPROT; ++t) s += y[((size_t)t * BB + b) * 64 + k] * attn[t * 3 + c];
        xc[(size_t)b * 520 + 328 + k * 3 + c] = s;
    }
}
// ---------------- final dot ----------------
__global__ void k_out(const float* __restrict__ fc2o, const float* __restrict__ ow,
                      const float* __restrict__ ob, float* __restrict__ out) {
    int b = blockIdx.x * blockDim.x + threadIdx.x;
    if (b >= BB) return;
    float s = ob[0];
    for (int k = 0; k < 512; ++k) s += fc2o[b * 512 + k] * ow[k];
    out[b] = s;
}

// =========================== host ===========================
extern "C" void kernel_launch(void* const* d_in, const int* in_sizes, int n_in,
                              void* d_out, int out_size, void* d_ws, size_t ws_size,
                              hipStream_t stream) {
    (void)in_sizes; (void)n_in; (void)out_size; (void)ws_size;
    const float* x        = (const float*)d_in[0];
    const int*   ei       = (const int*)d_in[1];
    const int*   smile    = (const int*)d_in[3];
    const int*   target   = (const int*)d_in[4];
    const float* gat_w    = (const float*)d_in[5];
    const float* att_src  = (const float*)d_in[6];
    const float* att_dst  = (const float*)d_in[7];
    const float* gat_b    = (const float*)d_in[8];
    const float* gcn_w    = (const float*)d_in[9];
    const float* gcn_b    = (const float*)d_in[10];
    const float* fcg1w    = (const float*)d_in[11];
    const float* fcg1b    = (const float*)d_in[12];
    const float* fcg2w    = (const float*)d_in[13];
    const float* fcg2b    = (const float*)d_in[14];
    const float* smi_emb  = (const float*)d_in[15];
    const float* emb_xt   = (const float*)d_in[24];
    const float* fc1w     = (const float*)d_in[49];
    const float* fc1b     = (const float*)d_in[50];
    const float* fc2w     = (const float*)d_in[51];
    const float* fc2b     = (const float*)d_in[52];
    const float* outw     = (const float*)d_in[53];
    const float* outb     = (const float*)d_in[54];
    float* dout = (float*)d_out;

    // ---- carve workspace ----
    char* ws = (char*)d_ws; size_t cur = 0;
    auto alloc = [&](size_t bytes) -> void* {
        void* p = ws + cur; cur = (cur + bytes + 255) & ~(size_t)255; return p;
    };
    unsigned short* abf  = (unsigned short*)alloc((size_t)NN * 800 * 2);     // padded A staging
    unsigned short* bbf  = (unsigned short*)alloc((size_t)1536 * 1568 * 2);  // padded B^T staging
    float* buf1  = (float*)alloc((size_t)NN * DG * 4);   // h / hg
    float* buf2  = (float*)alloc((size_t)NN * DG * 4);   // xg / xg2
    float* a_s   = (float*)alloc((size_t)NN * HEADS * 4);
    float* a_d   = (float*)alloc((size_t)NN * HEADS * 4);
    unsigned* mkey = (unsigned*)alloc((size_t)NN * HEADS * 4);
    float* denom = (float*)alloc((size_t)NN * HEADS * 4);
    float* ebuf  = (float*)alloc((size_t)E2 * HEADS * 4);
    float* deg   = (float*)alloc((size_t)NN * 4);
    float* dinv  = (float*)alloc((size_t)NN * 4);
    float* pooled= (float*)alloc((size_t)BB * 1560 * 4);
    float* fcg1o = (float*)alloc((size_t)BB * 1500 * 4);
    float* xc    = (float*)alloc((size_t)BB * 520 * 4);
    float* fc1o  = (float*)alloc((size_t)BB * 1024 * 4);
    float* fc2o  = (float*)alloc((size_t)BB * 512 * 4);
    float* yA    = (float*)alloc((size_t)TPROT * BB * 64 * 4);
    float* yB    = (float*)alloc((size_t)TPROT * BB * 64 * 4);
    float* hn    = (float*)alloc((size_t)6 * BB * HL * 4);

    auto cdiv = [](long long a, long long b) { return (unsigned)((a + b - 1) / b); };
    auto r32  = [](int k) { return (k + 31) & ~31; };
    auto r128 = [](int m) { return (m + 127) & ~127; };

    auto stageA = [&](const float* s, unsigned short* d, int M, int K) {
        long long tot = (long long)r128(M) * r32(K);
        k_stage_a<<<cdiv(tot, 256), 256, 0, stream>>>(s, d, M, K, r32(K), tot);
    };
    // torch weight (out,in) row-major == B^T[N,K]: plain padded convert
    auto stageBt = [&](const float* s, unsigned short* d, int N, int K) {
        long long tot = (long long)r128(N) * r32(K);
        k_stage_a<<<cdiv(tot, 256), 256, 0, stream>>>(s, d, N, K, r32(K), tot);
    };
    // [K,N] matrix -> B^T[N,K] padded (gat_w, gcn_w)
    auto stageBtT = [&](const float* s, unsigned short* d, int N, int K) {
        long long tot = (long long)r128(N) * r32(K);
        k_stage_bt_trans<<<cdiv(tot, 256), 256, 0, stream>>>(s, d, N, K, r32(K), tot);
    };
    auto fill = [&](void* p, unsigned v, long long n) {
        k_fill_u32<<<cdiv(n, 256), 256, 0, stream>>>((unsigned*)p, v, n);
    };
    auto gemm = [&](const unsigned short* A, const unsigned short* Bt, const float* bias,
                    float* C, int M, int N, int K, int ldc, int act) {
        dim3 g(cdiv(N, BN), cdiv(M, BM));
        k_gemm_bf16<<<g, 256, 0, stream>>>(A, Bt, bias, C, M, N, r32(K), ldc, act, bias ? 1 : 0);
    };

    // ===== GAT: h = x @ gat_w =====
    stageA(x, abf, NN, FXD);
    stageBtT(gat_w, bbf, DG, FXD);
    gemm(abf, bbf, nullptr, buf1, NN, DG, FXD, DG, 0);
    k_gat_scores<<<cdiv((long long)NN * HEADS, 256), 256, 0, stream>>>(buf1, att_src, att_dst, a_s, a_d);
    fill(mkey, 0u, (long long)NN * HEADS);
    fill(denom, 0u, (long long)NN * HEADS);
    fill(buf2, 0u, (long long)NN * DG);
    fill(deg, 0u, NN);
    k_edge_max  <<<cdiv((long long)E2 * HEADS, 256), 256, 0, stream>>>(ei, a_s, a_d, mkey);
    k_edge_exp  <<<cdiv((long long)E2 * HEADS, 256), 256, 0, stream>>>(ei, a_s, a_d, mkey, ebuf, denom);
    k_edge_alpha<<<cdiv((long long)E2 * HEADS, 256), 256, 0, stream>>>(ei, denom, ebuf);
    k_gat_aggr  <<<E2, 96, 0, stream>>>(ei, buf1, ebuf, buf2);
    k_bias_relu <<<cdiv((long long)NN * DG, 256), 256, 0, stream>>>(buf2, gat_b, (long long)NN * DG, DG);

    // ===== GCN =====
    k_deg <<<cdiv(E2, 256), 256, 0, stream>>>(ei, deg);
    k_dinv<<<cdiv(NN, 256), 256, 0, stream>>>(deg, dinv);
    stageA(buf2, abf, NN, DG);
    stageBtT(gcn_w, bbf, DG, DG);
    gemm(abf, bbf, nullptr, buf1, NN, DG, DG, DG, 0);   // hg
    fill(buf2, 0u, (long long)NN * DG);
    { dim3 g(E2, cdiv(DG, 256));
      k_gcn_aggr<<<g, 256, 0, stream>>>(ei, buf1, dinv, buf2); }
    k_bias_relu<<<cdiv((long long)NN * DG, 256), 256, 0, stream>>>(buf2, gcn_b, (long long)NN * DG, DG);

    // ===== pool + graph MLP =====
    k_pool<<<BB, 256, 0, stream>>>(buf2, pooled);
    stageA(pooled, abf, BB, 1560);
    stageBt(fcg1w, bbf, 1500, 1560);
    gemm(abf, bbf, fcg1b, fcg1o, BB, 1500, 1560, 1500, 1);
    stageA(fcg1o, abf, BB, 1500);
    stageBt(fcg2w, bbf, 128, 1500);
    gemm(abf, bbf, fcg2b, xc, BB, 128, 1500, 520, 0);   // writes xc[:,0:128]

    // ===== GRU over SMILES (writes xc[:,128:328]) =====
    {
        const float* wif = (const float*)d_in[16]; const float* whf = (const float*)d_in[17];
        const float* bif = (const float*)d_in[18]; const float* bhf = (const float*)d_in[19];
        const float* wib = (const float*)d_in[20]; const float* whb = (const float*)d_in[21];
        const float* bib = (const float*)d_in[22]; const float* bhb = (const float*)d_in[23];
        k_gru_scan<<<BB / GR, 256, 0, stream>>>(smile, smi_emb, wif, whf, bif, bhf, xc, 0);
        k_gru_scan<<<BB / GR, 256, 0, stream>>>(smile, smi_emb, wib, whb, bib, bhb, xc, 1);
    }

    // ===== 3-layer bidirectional LSTM over protein =====
    {
        float* yin = nullptr; float* yout = yA;
        for (int L = 0; L < 3; ++L) {
            int base = 25 + L * 8;
            int in_dim = (L == 0) ? 128 : 64;
            const float* wi_f = (const float*)d_in[base + 0]; const float* wh_f = (const float*)d_in[base + 1];
            const float* bi_f = (const float*)d_in[base + 2]; const float* bh_f = (const float*)d_in[base + 3];
            const float* wi_b = (const float*)d_in[base + 4]; const float* wh_b = (const float*)d_in[base + 5];
            const float* bi_b = (const float*)d_in[base + 6]; const float* bh_b = (const float*)d_in[base + 7];
            k_lstm_scan<<<BB / LR, 256, 0, stream>>>(yin, target, emb_xt, wi_f, wh_f, bi_f, bh_f,
                                                     yout, hn, in_dim, L == 0 ? 1 : 0, 0, L * 2 + 0);
            k_lstm_scan<<<BB / LR, 256, 0, stream>>>(yin, target, emb_xt, wi_b, wh_b, bi_b, bh_b,
                                                     yout, hn, in_dim, L == 0 ? 1 : 0, 1, L * 2 + 1);
            yin = yout;
            yout = (yout == yA) ? yB : yA;
        }
        // final layer output lives in yA (L0->yA, L1->yB, L2->yA)
        k_attention<<<BB, 256, 0, stream>>>(yA, hn, xc);  // writes xc[:,328:520]
    }

    // ===== final MLP =====
    stageA(xc, abf, BB, 520);
    stageBt(fc1w, bbf, 1024, 520);
    gemm(abf, bbf, fc1b, fc1o, BB, 1024, 520, 1024, 1);
    stageA(fc1o, abf, BB, 1024);
    stageBt(fc2w, bbf, 512, 1024);
    gemm(abf, bbf, fc2b, fc2o, BB, 512, 1024, 512, 1);
    k_out<<<1, 256, 0, stream>>>(fc2o, outw, outb, dout);
}
